// GaussianPointCloudRasterisation_63462436765751
// MI455X (gfx1250) — matrix operand order repaired and checked by reading, not compiled, not run on hardware
//
#include <hip/hip_runtime.h>
#include <hip/hip_bf16.h>
#include <math.h>

typedef __attribute__((ext_vector_type(16))) _Float16 v16h;
typedef __attribute__((ext_vector_type(8)))  _Float16 v8h;
typedef __attribute__((ext_vector_type(8)))  float    v8f;
typedef __attribute__((ext_vector_type(4)))  float    v4f;
typedef __attribute__((ext_vector_type(4)))  unsigned int v4u;
typedef __attribute__((ext_vector_type(8)))  int      v8i;
typedef __attribute__((ext_vector_type(4)))  int      v4i;

// Problem dimensions fixed by the reference setup.
constexpr int N_GAUSS = 256;
constexpr int IMG_H   = 256;
constexpr int IMG_W   = 256;
constexpr int NGROUP  = N_GAUSS / 16;   // 16 groups of 16 gaussians

constexpr float NEAR_PLANE  = 0.8f;
constexpr float FAR_PLANE   = 1000.0f;
constexpr float ALPHA_SKIP  = 1.0f / 255.0f;
constexpr float ALPHA_CLAMP = 0.99f;
constexpr float ACC_BREAK   = 0.9999f;
constexpr float TWO_PI      = 6.283185307179586f;

// workspace layout (contiguous, DMA'd as one 32 KB block):
//   [0, 8192)      : params, N_GAUSS x 8 floats {u,v,ia,ib, ic,amp,pad,pad}
//   [8192, 32768)  : A fragments, [group][channel][lane][8 halves] f16
constexpr unsigned PARAMS_BYTES = N_GAUSS * 8 * sizeof(float);     // 8 KB
constexpr unsigned STAGE_BYTES  = PARAMS_BYTES + NGROUP * 3 * 32 * 8 * 2; // 32 KB
constexpr unsigned STAGE_DWORDS = STAGE_BYTES / 4;                 // 8192

__device__ __forceinline__ float sigmoidf_(float x) {
  return 1.0f / (1.0f + __expf(-x));
}

__device__ __forceinline__ void sh_basis(float x, float y, float z, float* b) {
  float xx = x * x, yy = y * y, zz = z * z;
  b[0]  = 0.28209479177387814f;
  b[1]  = -0.4886025119029199f * y;
  b[2]  =  0.4886025119029199f * z;
  b[3]  = -0.4886025119029199f * x;
  b[4]  =  1.0925484305920792f * x * y;
  b[5]  = -1.0925484305920792f * y * z;
  b[6]  =  0.31539156525252005f * (2.f * zz - xx - yy);
  b[7]  = -1.0925484305920792f * x * z;
  b[8]  =  0.5462742152960396f * (xx - yy);
  b[9]  = -0.5900435899266435f * y * (3.f * xx - yy);
  b[10] =  2.890611442640554f  * x * y * z;
  b[11] = -0.4570457994644658f * y * (4.f * zz - xx - yy);
  b[12] =  0.3731763325901154f * z * (2.f * zz - 3.f * xx - 3.f * yy);
  b[13] = -0.4570457994644658f * x * (4.f * zz - xx - yy);
  b[14] =  1.445305721320277f  * z * (xx - yy);
  b[15] = -0.5900435899266435f * x * (xx - 3.f * yy);
}

// ---------------------------------------------------------------------------
// Kernel 1: project, depth-sort (O(N^2) rank, N=256), 2D covariance,
// and pack per-group WMMA A-fragments (f16 coeffs) into the workspace.
// One block of N_GAUSS threads.
// ---------------------------------------------------------------------------
__global__ void __launch_bounds__(N_GAUSS)
preprocess_kernel(const float* __restrict__ pc,
                  const float* __restrict__ feats,
                  const float* __restrict__ Kmat,
                  const float* __restrict__ Tmat,
                  float* __restrict__ params,
                  _Float16* __restrict__ coefA) {
  __shared__ float s_key[N_GAUSS];
  __shared__ int   s_ord[N_GAUSS];
  __shared__ float s_u[N_GAUSS], s_v[N_GAUSS];
  __shared__ float s_px[N_GAUSS], s_py[N_GAUSS], s_pz[N_GAUSS];
  __shared__ float s_inc[N_GAUSS];

  const int n = threadIdx.x;

  const float fx = Kmat[0], cx = Kmat[2], fy = Kmat[4], cy = Kmat[5];
  float R[3][3], t[3];
#pragma unroll
  for (int i = 0; i < 3; ++i) {
#pragma unroll
    for (int j = 0; j < 3; ++j) R[i][j] = Tmat[i * 4 + j];
    t[i] = Tmat[i * 4 + 3];
  }

  // project
  const float p0 = pc[n * 3 + 0], p1 = pc[n * 3 + 1], p2 = pc[n * 3 + 2];
  const float xc = R[0][0] * p0 + R[0][1] * p1 + R[0][2] * p2 + t[0];
  const float yc = R[1][0] * p0 + R[1][1] * p1 + R[1][2] * p2 + t[1];
  const float zc = R[2][0] * p0 + R[2][1] * p1 + R[2][2] * p2 + t[2];
  const float zs = fmaxf(zc, 1e-6f);
  const float u = (fx * xc + cx * zc) / zs;
  const float v = (fy * yc + cy * zc) / zs;
  const bool in_cam = (zc > NEAR_PLANE) && (zc < FAR_PLANE) &&
                      (u >= 0.f) && (u < (float)IMG_W) &&
                      (v >= 0.f) && (v < (float)IMG_H);

  s_key[n] = in_cam ? zc : 1e10f;
  s_u[n] = u;  s_v[n] = v;
  s_px[n] = xc; s_py[n] = yc; s_pz[n] = zc;
  s_inc[n] = in_cam ? 1.f : 0.f;
  __syncthreads();

  // stable rank sort
  const float ki = s_key[n];
  int rank = 0;
  for (int j = 0; j < N_GAUSS; ++j) {
    const float kj = s_key[j];
    rank += (kj < ki) || ((kj == ki) && (j < n));
  }
  s_ord[rank] = n;
  __syncthreads();

  // thread index == sorted position r
  const int r = threadIdx.x;
  const int i = s_ord[r];
  const float* f = feats + i * 56;

  // quaternion -> rotmat
  float qx = f[0], qy = f[1], qz = f[2], qw = f[3];
  const float qn = rsqrtf(qx * qx + qy * qy + qz * qz + qw * qw);
  qx *= qn; qy *= qn; qz *= qn; qw *= qn;
  float Rq[3][3];
  Rq[0][0] = 1.f - 2.f * (qy * qy + qz * qz);
  Rq[0][1] = 2.f * (qx * qy - qz * qw);
  Rq[0][2] = 2.f * (qx * qz + qy * qw);
  Rq[1][0] = 2.f * (qx * qy + qz * qw);
  Rq[1][1] = 1.f - 2.f * (qx * qx + qz * qz);
  Rq[1][2] = 2.f * (qy * qz - qx * qw);
  Rq[2][0] = 2.f * (qx * qz - qy * qw);
  Rq[2][1] = 2.f * (qy * qz + qx * qw);
  Rq[2][2] = 1.f - 2.f * (qx * qx + qy * qy);

  const float sc[3] = {__expf(f[4]), __expf(f[5]), __expf(f[6])};
  float M[3][3], Sig[3][3];
#pragma unroll
  for (int a = 0; a < 3; ++a)
#pragma unroll
    for (int b = 0; b < 3; ++b) M[a][b] = Rq[a][b] * sc[b];
#pragma unroll
  for (int a = 0; a < 3; ++a)
#pragma unroll
    for (int b = 0; b < 3; ++b) {
      float acc = 0.f;
#pragma unroll
      for (int k = 0; k < 3; ++k) acc += M[a][k] * M[b][k];
      Sig[a][b] = acc;
    }

  const float z = s_pz[i];
  const float iz = 1.f / z, iz2 = iz * iz;
  const float J00 = fx * iz, J02 = -fx * s_px[i] * iz2;
  const float J11 = fy * iz, J12 = -fy * s_py[i] * iz2;
  float JW0[3], JW1[3];
#pragma unroll
  for (int j = 0; j < 3; ++j) {
    JW0[j] = J00 * R[0][j] + J02 * R[2][j];
    JW1[j] = J11 * R[1][j] + J12 * R[2][j];
  }
  float t0[3], t1[3];
#pragma unroll
  for (int j = 0; j < 3; ++j) {
    t0[j] = JW0[0] * Sig[0][j] + JW0[1] * Sig[1][j] + JW0[2] * Sig[2][j];
    t1[j] = JW1[0] * Sig[0][j] + JW1[1] * Sig[1][j] + JW1[2] * Sig[2][j];
  }
  const float c00 = t0[0] * JW0[0] + t0[1] * JW0[1] + t0[2] * JW0[2];
  const float c01 = t0[0] * JW1[0] + t0[1] * JW1[1] + t0[2] * JW1[2];
  const float c11 = t1[0] * JW1[0] + t1[1] * JW1[1] + t1[2] * JW1[2];

  const float det = fmaxf(c00 * c11 - c01 * c01, 1e-12f);
  const float idet = 1.f / det;
  float ia = c11 * idet, ib = -c01 * idet, icv = c00 * idet;
  const float prefac = 1.f / (TWO_PI * sqrtf(det));
  const float opac = sigmoidf_(f[7]);
  float amp = prefac * opac;
  if (s_inc[i] == 0.f) { ia = 0.f; ib = 0.f; icv = 0.f; amp = 0.f; }

  float* pp = params + r * 8;
  pp[0] = s_u[i]; pp[1] = s_v[i];
  pp[2] = ia;     pp[3] = ib;
  pp[4] = icv;    pp[5] = amp;
  pp[6] = 0.f;    pp[7] = 0.f;

  // Pack A-fragments (16-bit A 16x32 layout, K padded 16->32 with zeros):
  // lanes 0-15 hold K=0..7 in halves 0..7; lanes 16-31 hold K=8..15.
  const int group = r >> 4, ml = r & 15;
#pragma unroll
  for (int c = 0; c < 3; ++c) {
#pragma unroll
    for (int half = 0; half < 2; ++half) {
      const int lane = ml + 16 * half;
      _Float16* dst = coefA + ((size_t)((group * 3 + c) * 32 + lane)) * 8;
#pragma unroll
      for (int kk = 0; kk < 8; ++kk)
        dst[kk] = (_Float16)f[8 + c * 16 + half * 8 + kk];
    }
  }
}

// ---------------------------------------------------------------------------
// TDM: DMA `STAGE_DWORDS` dwords from `gsrc` into LDS at byte offset
// `lds_off`. D# per CDNA5 ISA: 1-row tile, 4-byte
// elements, tensor_dim0 = tile_dim0 = STAGE_DWORDS.
// ---------------------------------------------------------------------------
__device__ __forceinline__ void tdm_stage_to_lds(const void* gsrc,
                                                 unsigned lds_off) {
  const unsigned long long ga = (unsigned long long)(uintptr_t)gsrc;
  v4u g0;
  g0[0] = 1u;                                           // count=1, user D#
  g0[1] = lds_off;                                      // lds_addr (bytes)
  g0[2] = (unsigned)(ga & 0xFFFFFFFFull);               // global_addr[31:0]
  g0[3] = (unsigned)((ga >> 32) & 0x01FFFFFFull)        // global_addr[56:32]
          | (2u << 30);                                 // type = 2 ("image")
  v8i g1;
  g1[0] = (int)(2u << 16);                              // data_size = 4B
  g1[1] = (int)((STAGE_DWORDS & 0xFFFFu) << 16);        // tensor_dim0 lo16
  g1[2] = (int)((STAGE_DWORDS >> 16) | (1u << 16));     // dim0 hi16 | dim1=1
  g1[3] = (int)((STAGE_DWORDS & 0xFFFFu) << 16);        // tile_dim0
  g1[4] = 1;                                            // tile_dim1=1, dim2=0
  g1[5] = (int)STAGE_DWORDS;                            // dim0_stride lo32
  g1[6] = 0;                                            // stride hi | dim1_stride
  g1[7] = 0;
  const v4i z4 = {0, 0, 0, 0};
#if defined(__clang_major__) && (__clang_major__ >= 23)
  const v8i z8 = {0, 0, 0, 0, 0, 0, 0, 0};
  __builtin_amdgcn_tensor_load_to_lds(g0, g1, z4, z4, z8, 0);
#else
  __builtin_amdgcn_tensor_load_to_lds(g0, g1, z4, z4, 0);
#endif
  __builtin_amdgcn_s_wait_tensorcnt(0);
}

// ---------------------------------------------------------------------------
// Kernel 2: raster. Wave 0 TDM-stages the 32 KB gaussian table (params +
// WMMA A-fragments) into LDS; each wave owns 16 pixels (WMMA columns).
// Per 16-gaussian depth group: 3x v_wmma_f32_16x16x32_f16 (one per channel)
// fed from LDS, lane-local alpha scan over 8 gaussians, half-wave shuffles
// stitch the cumulative-alpha chain and merge RGB.
// ---------------------------------------------------------------------------
__global__ void __launch_bounds__(256)
raster_kernel(const float* __restrict__ Kmat,
              const float* __restrict__ Tmat,
              const void* __restrict__ stage_src,   // params+coefA, 32 KB
              float* __restrict__ out) {
  __shared__ __align__(16) unsigned char s_stage[STAGE_BYTES];

  // Stage gaussian table via the Tensor Data Mover (one wave issues; TDM
  // ignores EXEC). Low 32 bits of the generic LDS address = LDS byte offset.
  if (threadIdx.x < 32) {
    tdm_stage_to_lds(stage_src, (unsigned)(uintptr_t)(void*)&s_stage[0]);
  }
  __syncthreads();

  const v4f* __restrict__ paramv = (const v4f*)&s_stage[0];
  const v8h* __restrict__ coefv  = (const v8h*)&s_stage[PARAMS_BYTES];

  const int lane = threadIdx.x & 31;
  const int wave = (blockIdx.x * blockDim.x + threadIdx.x) >> 5;
  const int col  = lane & 15;             // pixel column within the wave
  const int pix  = wave * 16 + col;       // linear pixel id
  const int px   = pix & (IMG_W - 1);
  const int py   = pix >> 8;              // IMG_W == 256

  const float fx = Kmat[0], cx = Kmat[2], fy = Kmat[4], cy = Kmat[5];
  float R[3][3];
#pragma unroll
  for (int i = 0; i < 3; ++i)
#pragma unroll
    for (int j = 0; j < 3; ++j) R[i][j] = Tmat[i * 4 + j];

  // view ray direction (world), matches pix @ Kinv.T @ R then normalize
  const float dcx = ((float)px - cx) / fx;
  const float dcy = ((float)py - cy) / fy;
  float dwx = dcx * R[0][0] + dcy * R[1][0] + R[2][0];
  float dwy = dcx * R[0][1] + dcy * R[1][1] + R[2][1];
  float dwz = dcx * R[0][2] + dcy * R[1][2] + R[2][2];
  const float rn = rsqrtf(dwx * dwx + dwy * dwy + dwz * dwz);
  dwx *= rn; dwy *= rn; dwz *= rn;

  float bas[16];
  sh_basis(dwx, dwy, dwz, bas);

  // B fragment (32x16, K padded): lanes 0-15 carry K=0..7, lanes 16-31 K=8..15 in halves 0..7; halves 8..15 zero.
  v16h bfrag = {};
  {
#pragma unroll
    for (int k = 0; k < 8; ++k) bfrag[k] = (_Float16)bas[((lane >> 4) << 3) + k];
  }

  const float pxc = (float)px + 0.5f;
  const float pyc = (float)py + 0.5f;

  float Cg = 0.f;                 // cumulative alpha entering this group
  float rr = 0.f, rg = 0.f, rb = 0.f;

  for (int g = 0; g < NGROUP; ++g) {
    // A fragments for the 3 channels of this gaussian group (from LDS)
    v16h af0 = {}, af1 = {}, af2 = {};
    {
      const v8h lo0 = coefv[(g * 3 + 0) * 32 + lane];
      const v8h lo1 = coefv[(g * 3 + 1) * 32 + lane];
      const v8h lo2 = coefv[(g * 3 + 2) * 32 + lane];
#pragma unroll
      for (int kk = 0; kk < 8; ++kk) {
        af0[kk] = lo0[kk]; af1[kk] = lo1[kk]; af2[kk] = lo2[kk];
      }
    }
    const v8f cz = {};
    // D: lane = pixel column, vgpr r = gaussian row (0-7 low half, 8-15 high)
    v8f d0 = __builtin_amdgcn_wmma_f32_16x16x32_f16(false, af0, false, bfrag,
                                                    (short)0, cz, false, false);
    v8f d1 = __builtin_amdgcn_wmma_f32_16x16x32_f16(false, af1, false, bfrag,
                                                    (short)0, cz, false, false);
    v8f d2 = __builtin_amdgcn_wmma_f32_16x16x32_f16(false, af2, false, bfrag,
                                                    (short)0, cz, false, false);

    // alphas for this lane's 8 gaussians (depth order), params from LDS
    const int mbase = g * 16 + ((lane >> 4) << 3);
    float a8[8];
    float S = 0.f;
#pragma unroll
    for (int r = 0; r < 8; ++r) {
      const v4f pA = paramv[(mbase + r) * 2 + 0];   // u, v, ia, ib
      const v4f pB = paramv[(mbase + r) * 2 + 1];   // ic, amp, -, -
      const float du = pA.x - pxc;
      const float dv = pA.y - pyc;
      const float quad = pA.z * du * du + pB.x * dv * dv + 2.f * pA.w * du * dv;
      const float araw = pB.y * __expf(-0.5f * quad);
      const float a = (araw < ALPHA_SKIP) ? 0.f : fminf(araw, ALPHA_CLAMP);
      a8[r] = a;
      S += a;
    }

    // stitch the sequential cumsum across the two half-waves
    const float S_lo = __shfl(S, col, 32);           // sum over gaussians 0..7
    float Crun = (lane < 16) ? Cg : (Cg + S_lo);

#pragma unroll
    for (int r = 0; r < 8; ++r) {
      const float a = a8[r];
      Crun += a;
      const float w = (Crun <= ACC_BREAK) ? a * (1.f - (Crun - a)) : 0.f;
      rr += w * sigmoidf_(d0[r]);
      rg += w * sigmoidf_(d1[r]);
      rb += w * sigmoidf_(d2[r]);
    }

    const float S_hi = __shfl(S, col + 16, 32);      // sum over gaussians 8..15
    Cg += S_lo + S_hi;
  }

  // merge the two half-wave partial colors for each pixel
  rr += __shfl_xor(rr, 16, 32);
  rg += __shfl_xor(rg, 16, 32);
  rb += __shfl_xor(rb, 16, 32);

  if (lane < 16) {
    float* o = out + (size_t)pix * 3;
    o[0] = rr; o[1] = rg; o[2] = rb;
  }
}

extern "C" void kernel_launch(void* const* d_in, const int* in_sizes, int n_in,
                              void* d_out, int out_size, void* d_ws, size_t ws_size,
                              hipStream_t stream) {
  (void)in_sizes; (void)n_in; (void)out_size; (void)ws_size;
  const float* pc    = (const float*)d_in[0];
  const float* feats = (const float*)d_in[1];
  const float* Kmat  = (const float*)d_in[2];
  const float* Tmat  = (const float*)d_in[3];
  float* out = (float*)d_out;

  float*    params = (float*)d_ws;
  _Float16* coefA  = (_Float16*)((char*)d_ws + PARAMS_BYTES);

  preprocess_kernel<<<1, N_GAUSS, 0, stream>>>(pc, feats, Kmat, Tmat, params, coefA);

  const int waves_total = (IMG_H * IMG_W) / 16;   // 4096 waves, 16 px each
  const int blocks      = waves_total / 8;        // 8 waves (256 thr) per block
  raster_kernel<<<blocks, 256, 0, stream>>>(Kmat, Tmat, d_ws, out);
}
